// MultiLayerAttention_86002425135538
// MI455X (gfx1250) — matrix operand order repaired and checked by reading, not compiled, not run on hardware
//
#include <hip/hip_runtime.h>
#include <hip/hip_bf16.h>
#include <stdint.h>

// Problem constants (from reference)
#define B_  2
#define S_  1024
#define H_  2048
#define NH_ 16
#define HD_ 128
#define L_  4

typedef __attribute__((ext_vector_type(16))) __bf16 bf16x16;
typedef __attribute__((ext_vector_type(8)))  float  fx8;

union Frag16 {
    bf16x16 v;
    unsigned short u[16];
    uint4 q[2];
};

__device__ __forceinline__ unsigned short f2bf(float f) {
    union { float f; unsigned u; } x; x.f = f;
    unsigned r = x.u + 0x7FFFu + ((x.u >> 16) & 1u);   // round-to-nearest-even
    return (unsigned short)(r >> 16);
}

__device__ __forceinline__ fx8 vzero8() {
    fx8 z = {0.f,0.f,0.f,0.f,0.f,0.f,0.f,0.f};
    return z;
}

// A fragment (16x32 bf16, MxK): lane&15 = M row; contiguous K-run of 16
// at offset 16*g (g = lane>>4), matching load_b16.
__device__ __forceinline__ void load_a16(Frag16& fa, const unsigned short* base,
                                         int lda, int row0, int k0, int lane) {
    int g = lane >> 4, m = lane & 15;
    const unsigned short* p = base + (size_t)(row0 + m) * lda + k0;
    fa.q[0] = *(const uint4*)(p + 16 * g);
    fa.q[1] = *(const uint4*)(p + 16 * g + 8);
}

// B fragment (32x16 bf16, KxN): lane&15 = N col; contiguous K-run of 16 at
// offset 16*g. Source is stored row-per-N (i.e. transposed weight / K rows /
// V^T rows) so the run is contiguous in memory.
__device__ __forceinline__ void load_b16(Frag16& fb, const unsigned short* base,
                                         int ld, int n0, int k0, int lane) {
    int g = lane >> 4, n = lane & 15;
    const unsigned short* p = base + (size_t)(n0 + n) * ld + k0 + 16 * g;
    fb.q[0] = *(const uint4*)(p);
    fb.q[1] = *(const uint4*)(p + 8);
}

__device__ __forceinline__ fx8 wmma_bf16(const Frag16& a, const Frag16& b, fx8 c) {
    return __builtin_amdgcn_wmma_f32_16x16x32_bf16(false, a.v, false, b.v,
                                                   (short)0, c, false, false);
}

// ---------------- small prep kernels ----------------

// W[k][n] (HxH fp32) -> WT[n][k] bf16
__global__ void k_transpose_bf16(const float* __restrict__ W,
                                 unsigned short* __restrict__ WT) {
    int i = blockIdx.x * blockDim.x + threadIdx.x;   // over H_*H_, n-major
    int n = i >> 11, k = i & (H_ - 1);
    WT[i] = f2bf(W[(size_t)k * H_ + n]);
}

__global__ void k_f32_to_bf16(const float* __restrict__ src,
                              unsigned short* __restrict__ dst, int n) {
    int i = (blockIdx.x * blockDim.x + threadIdx.x) * 4;
    if (i < n) {
        float4 f = *(const float4*)(src + i);
        dst[i + 0] = f2bf(f.x); dst[i + 1] = f2bf(f.y);
        dst[i + 2] = f2bf(f.z); dst[i + 3] = f2bf(f.w);
    }
}

// gate[b][l] = softmax_l( mean_s(hs[b,s,:]) @ Wg )
__global__ void k_gate(const float* __restrict__ hs, const float* __restrict__ Wg,
                       float* __restrict__ gate) {
    int b = blockIdx.x;
    float acc[L_] = {0.f, 0.f, 0.f, 0.f};
    for (int h = threadIdx.x; h < H_; h += blockDim.x) {
        const float* p = hs + (size_t)b * S_ * H_ + h;
        float m = 0.f;
        for (int s = 0; s < S_; s++) m += p[(size_t)s * H_];
        m *= (1.f / S_);
        #pragma unroll
        for (int l = 0; l < L_; l++) acc[l] += m * Wg[h * L_ + l];
    }
    __shared__ float red[256][L_];
    #pragma unroll
    for (int l = 0; l < L_; l++) red[threadIdx.x][l] = acc[l];
    __syncthreads();
    for (int off = 128; off > 0; off >>= 1) {
        if ((int)threadIdx.x < off)
            #pragma unroll
            for (int l = 0; l < L_; l++) red[threadIdx.x][l] += red[threadIdx.x + off][l];
        __syncthreads();
    }
    if (threadIdx.x == 0) {
        float mx = red[0][0];
        #pragma unroll
        for (int l = 1; l < L_; l++) mx = fmaxf(mx, red[0][l]);
        float e[L_], s = 0.f;
        #pragma unroll
        for (int l = 0; l < L_; l++) { e[l] = __expf(red[0][l] - mx); s += e[l]; }
        #pragma unroll
        for (int l = 0; l < L_; l++) gate[b * L_ + l] = e[l] / s;
    }
}

// ---------------- projection GEMMs (WMMA) ----------------

// Q/K projection + RoPE. One wave computes TWO rope tile pairs of one head:
// columns (d0, d0+16) and their rotate-half partners (d0+64, d0+80), all four
// tiles sharing one A fragment per k-step. All 4 B fragments have distinct
// live ranges so their loads overlap the WMMAs (partial loadcnt waits).
// out layout: [(Mrows/S)*NH, S, HD] bf16.
__global__ void __launch_bounds__(256)
k_proj_rope(const unsigned short* __restrict__ Abf,   // [Mrows, H] bf16
            const unsigned short* __restrict__ WT,    // [H(n), H(k)] bf16
            const float* __restrict__ bias,           // [H]
            unsigned short* __restrict__ out,
            int Mrows, float scale) {
    int lane = threadIdx.x & 31;
    int t = blockIdx.x * (blockDim.x >> 5) + (threadIdx.x >> 5);
    int nColT = NH_ * 2;                          // 2 pair-groups per head
    int nTasks = (Mrows / 16) * nColT;
    if (t >= nTasks) return;
    int ct = t % nColT, rt = t / nColT;
    int head = ct >> 1;
    int d0 = (ct & 1) * 32;                       // 0 or 32
    int c0 = head * HD_ + d0;                     // x1 tiles at c0, c0+16
    int row0 = rt * 16;                           // x2 tiles at c0+64, c0+80

    fx8 acc[4];                                   // 0,1: x1 pair; 2,3: x2 pair
    #pragma unroll
    for (int q = 0; q < 4; q++) acc[q] = vzero8();

    Frag16 fa, fb[4];
    for (int k0 = 0; k0 < H_; k0 += 32) {
        load_a16(fa, Abf, H_, row0, k0, lane);
        load_b16(fb[0], WT, H_, c0,      k0, lane);
        load_b16(fb[1], WT, H_, c0 + 16, k0, lane);
        load_b16(fb[2], WT, H_, c0 + 64, k0, lane);
        load_b16(fb[3], WT, H_, c0 + 80, k0, lane);
        if (k0 + 32 < H_)
            __builtin_prefetch(Abf + (size_t)(row0 + (lane & 15)) * H_ + k0 + 32, 0, 3);
        #pragma unroll
        for (int q = 0; q < 4; q++) acc[q] = wmma_bf16(fa, fb[q], acc[q]);
    }

    int g = lane >> 4, n = lane & 15;
    int lb = row0 / S_;
    size_t obase = ((size_t)lb * NH_ + head) * S_ * HD_;
    #pragma unroll
    for (int p = 0; p < 2; p++) {
        int dcol = d0 + 16 * p;                   // within-head col of x1 tile
        float b1 = bias[c0 + 16 * p + n];
        float b2 = bias[c0 + 64 + 16 * p + n];
        int j = dcol + n;                         // rope freq index, 0..63
        float invf = __expf(-(float)j * 0.14391157f);  // 10000^(-j/64)
        #pragma unroll
        for (int r = 0; r < 8; r++) {
            int row = row0 + r + 8 * g;
            int s = row & (S_ - 1);
            float ang = (float)s * invf;
            float sn, c;
            __sincosf(ang, &sn, &c);
            float x1 = acc[p][r] + b1;
            float x2 = acc[2 + p][r] + b2;
            float o1 = (x1 * c - x2 * sn) * scale;
            float o2 = (x2 * c + x1 * sn) * scale;
            out[obase + (size_t)s * HD_ + dcol + n]      = f2bf(o1);
            out[obase + (size_t)s * HD_ + dcol + 64 + n] = f2bf(o2);
        }
    }
}

// V projection, 16x64 output per wave (4 B tiles share one A fragment),
// stored transposed: out [(Mrows/S)*NH, HD, S] bf16
__global__ void __launch_bounds__(256)
k_proj_v(const unsigned short* __restrict__ Abf, const unsigned short* __restrict__ WT,
         const float* __restrict__ bias, unsigned short* __restrict__ outT, int Mrows) {
    int lane = threadIdx.x & 31;
    int t = blockIdx.x * (blockDim.x >> 5) + (threadIdx.x >> 5);
    int nColT = H_ / 64;                          // 32
    int nTasks = (Mrows / 16) * nColT;
    if (t >= nTasks) return;
    int ct = t % nColT, rt = t / nColT;
    int n0 = ct * 64, row0 = rt * 16;

    fx8 acc[4];
    #pragma unroll
    for (int q = 0; q < 4; q++) acc[q] = vzero8();
    Frag16 fa, fb[4];
    for (int k0 = 0; k0 < H_; k0 += 32) {
        load_a16(fa, Abf, H_, row0, k0, lane);
        #pragma unroll
        for (int q = 0; q < 4; q++) load_b16(fb[q], WT, H_, n0 + 16 * q, k0, lane);
        if (k0 + 32 < H_)
            __builtin_prefetch(Abf + (size_t)(row0 + (lane & 15)) * H_ + k0 + 32, 0, 3);
        #pragma unroll
        for (int q = 0; q < 4; q++) acc[q] = wmma_bf16(fa, fb[q], acc[q]);
    }
    int g = lane >> 4, n = lane & 15;
    int lb = row0 / S_;
    #pragma unroll
    for (int q = 0; q < 4; q++) {
        int col = n0 + 16 * q + n;
        int head = col >> 7, d = col & 127;
        float bv = bias[col];
        size_t base = (((size_t)lb * NH_ + head) * HD_ + d) * S_;
        #pragma unroll
        for (int r = 0; r < 8; r++) {
            int row = row0 + r + 8 * g;
            int s = row & (S_ - 1);
            outT[base + s] = f2bf(acc[q][r] + bv);
        }
    }
}

// ---------------- attention (flash-style, gate-fused) ----------------

#define AW 4   // waves per block
__global__ void __launch_bounds__(128)
k_attn(const unsigned short* __restrict__ Qb,   // [B*NH, S, HD]  (pre-scaled 1/sqrt(HD))
       const unsigned short* __restrict__ Kb,   // [L*B*NH, S, HD]
       const unsigned short* __restrict__ Vt,   // [L*B*NH, HD, S]
       const float* __restrict__ gate,          // [B, L]
       unsigned short* __restrict__ mrg) {      // [B, S, NH*HD]
    __shared__ __align__(16) unsigned short Pl[AW][16 * 32];
    int lane = threadIdx.x & 31;
    int w = threadIdx.x >> 5;
    int t = blockIdx.x * AW + w;
    int nQt = S_ / 16;
    if (t >= B_ * NH_ * nQt) return;
    int qt = t % nQt, bh = t / nQt;
    int h = bh % NH_, b = bh / NH_;
    unsigned short* Pw = &Pl[w][0];
    int g = lane >> 4, n = lane & 15;

    Frag16 qf[4];
    const unsigned short* Qbase = Qb + ((size_t)(b * NH_ + h)) * S_ * HD_;
    #pragma unroll
    for (int kk = 0; kk < 4; kk++) load_a16(qf[kk], Qbase, HD_, qt * 16, kk * 32, lane);

    fx8 macc[8];
    #pragma unroll
    for (int c = 0; c < 8; c++) macc[c] = vzero8();

    for (int l = 0; l < L_; l++) {
        const unsigned short* Kbase = Kb + ((size_t)((l * B_ + b) * NH_ + h)) * S_ * HD_;
        const unsigned short* Vbase = Vt + ((size_t)((l * B_ + b) * NH_ + h)) * HD_ * S_;
        fx8 oacc[8];
        #pragma unroll
        for (int c = 0; c < 8; c++) oacc[c] = vzero8();
        float mrow[8], lrow[8];
        #pragma unroll
        for (int r = 0; r < 8; r++) { mrow[r] = -3.0e38f; lrow[r] = 0.f; }

        for (int j0 = 0; j0 < S_; j0 += 32) {
            fx8 s0 = vzero8(), s1 = vzero8();
            Frag16 kb0, kb1;
            #pragma unroll
            for (int kk = 0; kk < 4; kk++) {
                load_b16(kb0, Kbase, HD_, j0,      kk * 32, lane);  // keys j0..+15
                load_b16(kb1, Kbase, HD_, j0 + 16, kk * 32, lane);  // keys j0+16..+31
                s0 = wmma_bf16(qf[kk], kb0, s0);
                s1 = wmma_bf16(qf[kk], kb1, s1);
            }
            // online softmax over this 32-key slab; row stats replicated per 16-lane half
            #pragma unroll
            for (int r = 0; r < 8; r++) {
                float a0 = s0[r], a1 = s1[r];
                float mx = fmaxf(a0, a1);
                #pragma unroll
                for (int off = 1; off < 16; off <<= 1)
                    mx = fmaxf(mx, __shfl_xor(mx, off, 32));
                float mn = fmaxf(mrow[r], mx);
                float p0 = __expf(a0 - mn), p1 = __expf(a1 - mn);
                float rsum = p0 + p1;
                #pragma unroll
                for (int off = 1; off < 16; off <<= 1)
                    rsum += __shfl_xor(rsum, off, 32);
                float alpha = __expf(mrow[r] - mn);
                lrow[r] = lrow[r] * alpha + rsum;
                mrow[r] = mn;
                #pragma unroll
                for (int c = 0; c < 8; c++) oacc[c][r] *= alpha;
                int rowi = r + 8 * g;
                Pw[rowi * 32 + n]      = f2bf(p0);   // C-layout -> LDS row-major
                Pw[rowi * 32 + 16 + n] = f2bf(p1);
            }
            // intra-wave LDS store->load hazard: DS ops are in-order per wave;
            // fence the compiler and drain DScnt before the transposed read.
            asm volatile("s_wait_dscnt 0" ::: "memory");
            Frag16 pf;
            load_a16(pf, Pw, 32, 0, 0, lane);        // P as A-fragment (16x32)
            Frag16 vb0, vb1;
            #pragma unroll
            for (int c = 0; c < 8; c += 2) {         // pairwise: overlap load+WMMA
                load_b16(vb0, Vbase, S_, (c + 0) * 16, j0, lane);
                load_b16(vb1, Vbase, S_, (c + 1) * 16, j0, lane);
                oacc[c + 0] = wmma_bf16(pf, vb0, oacc[c + 0]);
                oacc[c + 1] = wmma_bf16(pf, vb1, oacc[c + 1]);
            }
        }
        float wl = gate[b * L_ + l];
        #pragma unroll
        for (int r = 0; r < 8; r++) {
            float sc = wl / lrow[r];
            #pragma unroll
            for (int c = 0; c < 8; c++) macc[c][r] += oacc[c][r] * sc;
        }
    }
    // store merged heads: [B, S, NH*HD] bf16
    #pragma unroll
    for (int c = 0; c < 8; c++)
        #pragma unroll
        for (int r = 0; r < 8; r++) {
            int row = qt * 16 + r + 8 * g;
            size_t addr = ((size_t)b * S_ + row) * (NH_ * HD_) + h * HD_ + c * 16 + n;
            mrg[addr] = f2bf(macc[c][r]);
        }
}

// ---------------- output projection (16x64 per wave) ----------------

__global__ void __launch_bounds__(256)
k_out_gemm(const unsigned short* __restrict__ Mrg, const unsigned short* __restrict__ WoT,
           const float* __restrict__ bo, float* __restrict__ out) {
    int lane = threadIdx.x & 31;
    int t = blockIdx.x * (blockDim.x >> 5) + (threadIdx.x >> 5);
    int nColT = H_ / 64;                          // 32
    if (t >= (B_ * S_ / 16) * nColT) return;
    int ct = t % nColT, rt = t / nColT;
    int n0 = ct * 64, row0 = rt * 16;

    fx8 acc[4];
    #pragma unroll
    for (int q = 0; q < 4; q++) acc[q] = vzero8();
    Frag16 fa, fb[4];
    for (int k0 = 0; k0 < H_; k0 += 32) {
        load_a16(fa, Mrg, H_, row0, k0, lane);
        #pragma unroll
        for (int q = 0; q < 4; q++) load_b16(fb[q], WoT, H_, n0 + 16 * q, k0, lane);
        if (k0 + 32 < H_)
            __builtin_prefetch(Mrg + (size_t)(row0 + (lane & 15)) * H_ + k0 + 32, 0, 3);
        #pragma unroll
        for (int q = 0; q < 4; q++) acc[q] = wmma_bf16(fa, fb[q], acc[q]);
    }
    int g = lane >> 4, n = lane & 15;
    #pragma unroll
    for (int q = 0; q < 4; q++) {
        int col = n0 + 16 * q + n;
        float bv = bo[col];
        #pragma unroll
        for (int r = 0; r < 8; r++) {
            int row = row0 + r + 8 * g;
            out[(size_t)row * H_ + col] = acc[q][r] + bv;
        }
    }
}

// ---------------- host launcher ----------------

extern "C" void kernel_launch(void* const* d_in, const int* in_sizes, int n_in,
                              void* d_out, int out_size, void* d_ws, size_t ws_size,
                              hipStream_t stream) {
    const float* hs   = (const float*)d_in[0];
    const float* prev = (const float*)d_in[1];
    const float* Wq = (const float*)d_in[2];
    const float* bq = (const float*)d_in[3];
    const float* Wk = (const float*)d_in[4];
    const float* bk = (const float*)d_in[5];
    const float* Wv = (const float*)d_in[6];
    const float* bv = (const float*)d_in[7];
    const float* Wo = (const float*)d_in[8];
    const float* bo = (const float*)d_in[9];
    const float* Wg = (const float*)d_in[10];
    float* out = (float*)d_out;

    char* p = (char*)d_ws;
    auto carve = [&](size_t bytes) -> char* {
        char* r = p; p += (bytes + 255) & ~(size_t)255; return r;
    };
    const size_t WW = (size_t)H_ * H_;              // 4M elems
    unsigned short* WqT = (unsigned short*)carve(WW * 2);
    unsigned short* WkT = (unsigned short*)carve(WW * 2);
    unsigned short* WvT = (unsigned short*)carve(WW * 2);
    unsigned short* WoT = (unsigned short*)carve(WW * 2);
    const size_t NH1 = (size_t)B_ * S_ * H_;        // 4M
    const size_t NHL = (size_t)L_ * B_ * S_ * H_;   // 16M
    unsigned short* Hbf = (unsigned short*)carve(NH1 * 2);
    unsigned short* Pbf = (unsigned short*)carve(NHL * 2);
    unsigned short* Qb  = (unsigned short*)carve(NH1 * 2);
    unsigned short* Kb  = (unsigned short*)carve(NHL * 2);
    unsigned short* Vt  = (unsigned short*)carve(NHL * 2);
    unsigned short* Mrg = (unsigned short*)carve(NH1 * 2);
    float* gate = (float*)carve(B_ * L_ * sizeof(float));

    // 1) weights -> transposed bf16
    k_transpose_bf16<<<WW / 256, 256, 0, stream>>>(Wq, WqT);
    k_transpose_bf16<<<WW / 256, 256, 0, stream>>>(Wk, WkT);
    k_transpose_bf16<<<WW / 256, 256, 0, stream>>>(Wv, WvT);
    k_transpose_bf16<<<WW / 256, 256, 0, stream>>>(Wo, WoT);
    // 2) activations -> bf16
    k_f32_to_bf16<<<(unsigned)(NH1 / 1024), 256, 0, stream>>>(hs, Hbf, (int)NH1);
    k_f32_to_bf16<<<(unsigned)(NHL / 1024), 256, 0, stream>>>(prev, Pbf, (int)NHL);
    // 3) gate weights
    k_gate<<<B_, 256, 0, stream>>>(hs, Wg, gate);
    // 4) Q projection + RoPE, pre-scaled by 1/sqrt(HD)
    {
        int tasks = (B_ * S_ / 16) * (NH_ * 2);      // 4096 waves
        k_proj_rope<<<tasks / 8, 256, 0, stream>>>(Hbf, WqT, bq, Qb, B_ * S_,
                                                   0.08838834764f);
    }
    // 5) K projection + RoPE
    {
        int tasks = (L_ * B_ * S_ / 16) * (NH_ * 2); // 16384 waves
        k_proj_rope<<<tasks / 8, 256, 0, stream>>>(Pbf, WkT, bk, Kb, L_ * B_ * S_, 1.0f);
    }
    // 6) V projection (transposed store, 16x64 per wave)
    {
        int tasks = (L_ * B_ * S_ / 16) * (H_ / 64); // 16384 waves
        k_proj_v<<<tasks / 8, 256, 0, stream>>>(Pbf, WvT, bv, Vt, L_ * B_ * S_);
    }
    // 7) attention + gate-weighted merge
    {
        int tasks = B_ * NH_ * (S_ / 16);            // 2048 waves
        k_attn<<<tasks / AW, 32 * AW, 0, stream>>>(Qb, Kb, Vt, gate, Mrg);
    }
    // 8) output projection (16x64 per wave)
    {
        int tasks = (B_ * S_ / 16) * (H_ / 64);      // 4096 waves
        k_out_gemm<<<tasks / 8, 256, 0, stream>>>(Mrg, WoT, bo, out);
    }
}
